// AdjacencyDecoder_61186104099315
// MI455X (gfx1250) — matrix operand order repaired and checked by reading, not compiled, not run on hardware
//
#include <hip/hip_runtime.h>

typedef __attribute__((ext_vector_type(16))) __bf16 v16bf;
typedef __attribute__((ext_vector_type(8)))  float  v8f;

#define LATENT 128
#define NH1 64
#define NH2 32
#define WAVES 8
#define TILE_M 16
#define EDGES_PER_BLOCK (WAVES * TILE_M)

__global__ __launch_bounds__(256)
void edge_mlp_wmma(const float* __restrict__ z,
                   const long long* __restrict__ ei,   // [2, E] int64
                   const float* __restrict__ W1, const float* __restrict__ b1,
                   const float* __restrict__ W2, const float* __restrict__ b2,
                   const float* __restrict__ W3, const float* __restrict__ b3,
                   float* __restrict__ out, int n_edges)
{
    // W1 transposed [n][k] in bf16: B-fragment loads become contiguous LDS reads.
    __shared__ __bf16 sW1T[NH1 * (2 * LATENT)];   // 32 KB
    __shared__ __bf16 sW2T[NH2 * NH1];            // 4 KB
    __shared__ float4 scratch[WAVES][128];        // 2 KB / wave (h1 bf16 16x64 aliases h2 f32 16x32)

    const int tid  = threadIdx.x;
    const int lane = tid & 31;
    const int wave = tid >> 5;

    // ---- stage weights (fp32 -> bf16, transpose) ----
    for (int i = tid; i < 2 * LATENT * NH1; i += 256) {
        int k = i / NH1, n = i % NH1;
        sW1T[n * (2 * LATENT) + k] = (__bf16)W1[i];
    }
    for (int i = tid; i < NH1 * NH2; i += 256) {
        int k = i / NH2, n = i % NH2;
        sW2T[n * NH1 + k] = (__bf16)W2[i];
    }
    __syncthreads();

    const int m      = lane & 15;          // row / column within 16-half
    const int koff   = (lane >> 4) << 3;   // A-frag K base: 0 or 8
    const int kboff  = (lane >> 4) << 4;   // unused; B frags also start at koff
    const int rowoff = (lane >> 4) << 3;   // C rows: 0 or 8

    const int tile_base = blockIdx.x * EDGES_PER_BLOCK + wave * TILE_M;
    int e = tile_base + m;
    if (e >= n_edges) e = n_edges - 1;     // clamp: keep EXEC all-ones for WMMA
    const long long sn = ei[e];
    const long long dn = ei[n_edges + e];
    const float* zs = z + sn * LATENT;     // L2-resident gather
    const float* zd = z + dn * LATENT;

    // ---- layer 1: [16 x 256] @ [256 x 64], bf16 WMMA, f32 accum ----
    v8f acc[4] = {};
#pragma unroll
    for (int ks = 0; ks < 8; ++ks) {
        const int k0 = ks * 32;
        v16bf a;
        {   // K = k0+koff .. +7  (never crosses the src/dst concat boundary)
            int f = k0 + koff;
            const float* p = (f < LATENT) ? (zs + f) : (zd + (f - LATENT));
            float4 u = *(const float4*)p;
            float4 w = *(const float4*)(p + 4);
            a[0]=(__bf16)u.x; a[1]=(__bf16)u.y; a[2]=(__bf16)u.z; a[3]=(__bf16)u.w;
            a[4]=(__bf16)w.x; a[5]=(__bf16)w.y; a[6]=(__bf16)w.z; a[7]=(__bf16)w.w;
        }
        {   // K = k0+16+koff .. +7
            int f = k0 + 16 + koff;
            const float* p = (f < LATENT) ? (zs + f) : (zd + (f - LATENT));
            float4 u = *(const float4*)p;
            float4 w = *(const float4*)(p + 4);
            a[8]=(__bf16)u.x;  a[9]=(__bf16)u.y;  a[10]=(__bf16)u.z; a[11]=(__bf16)u.w;
            a[12]=(__bf16)w.x; a[13]=(__bf16)w.y; a[14]=(__bf16)w.z; a[15]=(__bf16)w.w;
        }
#pragma unroll
        for (int t = 0; t < 4; ++t) {
            const __bf16* bp = sW1T + (t * 16 + m) * (2 * LATENT) + k0 + koff;
            v16bf b;
            *(uint4*)&b       = *(const uint4*)bp;        // K = k0+koff .. +7, then k0+16+koff .. +7
            *((uint4*)&b + 1) = *(const uint4*)(bp + 16);
            acc[t] = __builtin_amdgcn_wmma_f32_16x16x32_bf16(
                false, a, false, b, (short)0, acc[t], false, false);
        }
    }

    // ---- bias + ReLU, store h1 (bf16) in [M][K] order for layer-2 A frags ----
    __bf16* h1w = (__bf16*)&scratch[wave][0];
#pragma unroll
    for (int t = 0; t < 4; ++t) {
        float bias = b1[t * 16 + m];
#pragma unroll
        for (int r = 0; r < 8; ++r) {
            float v = acc[t][r] + bias;
            v = v > 0.f ? v : 0.f;
            h1w[(rowoff + r) * NH1 + (t * 16 + m)] = (__bf16)v;
        }
    }
    asm volatile("s_wait_dscnt 0" ::: "memory");

    // ---- layer 2: [16 x 64] @ [64 x 32] ----
    v8f acc2[2] = {};
#pragma unroll
    for (int ks = 0; ks < 2; ++ks) {
        const int k0 = ks * 32;
        v16bf a;
        const __bf16* p0 = h1w + m * NH1 + k0 + koff;       // 8 bf16 = 16 B
        const __bf16* p1 = h1w + m * NH1 + k0 + 16 + koff;
        *(uint4*)&a       = *(const uint4*)p0;
        *((uint4*)&a + 1) = *(const uint4*)p1;
#pragma unroll
        for (int t = 0; t < 2; ++t) {
            const __bf16* bp = sW2T + (t * 16 + m) * NH1 + k0 + koff;
            v16bf b;
            *(uint4*)&b       = *(const uint4*)bp;
            *((uint4*)&b + 1) = *(const uint4*)(bp + 16);
            acc2[t] = __builtin_amdgcn_wmma_f32_16x16x32_bf16(
                false, a, false, b, (short)0, acc2[t], false, false);
        }
    }

    // ---- bias + ReLU, store h2 (f32) ; aliases h1 buffer (all h1 reads already issued) ----
    float* h2w = (float*)&scratch[wave][0];
#pragma unroll
    for (int t = 0; t < 2; ++t) {
        float bias = b2[t * 16 + m];
#pragma unroll
        for (int r = 0; r < 8; ++r) {
            float v = acc2[t][r] + bias;
            v = v > 0.f ? v : 0.f;
            h2w[(rowoff + r) * NH2 + (t * 16 + m)] = v;
        }
    }
    asm volatile("s_wait_dscnt 0" ::: "memory");

    // ---- layer 3: [16 x 32] @ [32 x 1] as per-lane dot (WMMA would waste 15/16 of tile) ----
    if (lane < 16) {
        int ee = tile_base + lane;
        if (ee < n_edges) {
            float s = b3[0];
#pragma unroll
            for (int n = 0; n < NH2; ++n) s += h2w[lane * NH2 + n] * W3[n];
            out[ee] = s;
        }
    }
}

extern "C" void kernel_launch(void* const* d_in, const int* in_sizes, int n_in,
                              void* d_out, int out_size, void* d_ws, size_t ws_size,
                              hipStream_t stream) {
    const float*     z  = (const float*)d_in[0];
    const long long* ei = (const long long*)d_in[1];
    const float*     W1 = (const float*)d_in[2];
    const float*     b1 = (const float*)d_in[3];
    const float*     W2 = (const float*)d_in[4];
    const float*     b2 = (const float*)d_in[5];
    const float*     W3 = (const float*)d_in[6];
    const float*     b3 = (const float*)d_in[7];
    float* out = (float*)d_out;

    int n_edges = in_sizes[1] / 2;
    int blocks  = (n_edges + EDGES_PER_BLOCK - 1) / EDGES_PER_BLOCK;
    hipLaunchKernelGGL(edge_mlp_wmma, dim3(blocks), dim3(256), 0, stream,
                       z, ei, W1, b1, W2, b2, W3, b3, out, n_edges);
}
